// BCMGRU_60103772340464
// MI455X (gfx1250) — matrix operand order repaired and checked by reading, not compiled, not run on hardware
//
#include <hip/hip_runtime.h>
#include <math.h>
#include <stdint.h>

typedef __bf16 bf16;
typedef __attribute__((ext_vector_type(8)))  bf16  v8bf;
typedef __attribute__((ext_vector_type(16))) bf16  v16bf;
typedef __attribute__((ext_vector_type(8)))  float v8f;
typedef int v4i_vs __attribute__((vector_size(4 * sizeof(int))));

#define T_STEPS 800
#define BATCH   32
#define DIM     512
#define G3      1536
#define NROWS   (T_STEPS * BATCH)   // 25600
#define EPS_F   1e-6f

#define NWG_REC 48                  // 48 WGs x 32 cols = 1536 cols of u
#define THR_REC 128                 // 4 waves: (mt,nt) in {0,1}x{0,1}

// ---- workspace layout (bytes, 256-aligned) ----
#define OFF_WOUT 0ull                                   // 25600*1536*4 = 157286400
#define OFF_XN   157286400ull                           // 25600*512*2  =  26214400
#define OFF_WIH  183500800ull                           // 1536*512*2   =   1572864
#define OFF_WHH  185073664ull                           // 1536*512*2   =   1572864
#define OFF_U    186646528ull                           // 2*32*1536*4  =    393216
#define OFF_BAR  187039744ull                           // barrier counter

static __device__ inline v8f wmma_bf16(v16bf a, v16bf b, v8f c) {
  // v_wmma_f32_16x16x32_bf16: D = A(16x32) * B(32x16) + C(16x16 f32)
  return __builtin_amdgcn_wmma_f32_16x16x32_bf16(false, a, false, b, (short)0, c, false, false);
}

static __device__ inline v16bf concat8(v8bf lo, v8bf hi) {
  return __builtin_shufflevector(lo, hi, 0,1,2,3,4,5,6,7,8,9,10,11,12,13,14,15);
}

// ---- gfx1250 async global->LDS copy (16B per lane), guarded fallback ----
static __device__ inline void lds_copy16(const bf16* gsrc, bf16* ldst) {
#if __has_builtin(__builtin_amdgcn_global_load_async_to_lds_b128)
  __builtin_amdgcn_global_load_async_to_lds_b128(
      (__attribute__((address_space(1))) v4i_vs*)(uintptr_t)gsrc,
      (__attribute__((address_space(3))) v4i_vs*)(unsigned)(uintptr_t)ldst,
      0, 0);
#else
  *(uint4*)ldst = *(const uint4*)gsrc;   // sync fallback: load + ds_store
#endif
}

static __device__ inline void lds_copy_fence() {
#if __has_builtin(__builtin_amdgcn_global_load_async_to_lds_b128)
#if __has_builtin(__builtin_amdgcn_s_wait_asynccnt)
  __builtin_amdgcn_s_wait_asynccnt(0);
#else
  asm volatile("s_wait_asynccnt 0x0" ::: "memory");
#endif
#endif
}

// ---------------- kernel 0: reset grid barrier ----------------
__global__ void bar_init_kernel(unsigned* bar) { *bar = 0u; }

// ---------------- kernel 1: LayerNorm(x) -> bf16 row-major ----------------
// one wave per row of 512; 8 rows per 256-thread block
__global__ void ln_x_kernel(const float* __restrict__ x,
                            const float* __restrict__ g0,
                            const float* __restrict__ b0,
                            bf16* __restrict__ xn) {
  const int row  = blockIdx.x * 8 + (threadIdx.x >> 5);
  const int lane = threadIdx.x & 31;
  const float* xr = x + (size_t)row * DIM;

  float v[16];
  float s = 0.f, ss = 0.f;
#pragma unroll
  for (int i = 0; i < 16; ++i) {
    float t = xr[lane + 32 * i];
    v[i] = t; s += t; ss += t * t;
  }
#pragma unroll
  for (int m = 1; m < 32; m <<= 1) {
    s  += __shfl_xor(s,  m);
    ss += __shfl_xor(ss, m);
  }
  const float mean = s * (1.0f / DIM);
  float var = (ss - (float)DIM * mean * mean) * (1.0f / (DIM - 1));
  var = fmaxf(var, 0.f);
  const float scale = 1.0f / (sqrtf(var) + EPS_F);

  bf16* orow = xn + (size_t)row * DIM;
#pragma unroll
  for (int i = 0; i < 16; ++i) {
    const int c = lane + 32 * i;
    orow[c] = (bf16)(g0[c] * (v[i] - mean) * scale + b0[c]);
  }
}

// ---------------- kernel 2: gather weights, transposed, -> bf16 ----------------
// idx is [K=512][N=1536]; output wT is [N][K] so each WMMA B-fragment
// (16 contiguous K per lane) is one contiguous 32B load.
__global__ void gather_wT_kernel(const float* __restrict__ vec,
                                 const int*   __restrict__ idx,
                                 bf16* __restrict__ wT) {
  const int id = blockIdx.x * 256 + threadIdx.x;   // over N*K = 786432
  if (id >= G3 * DIM) return;
  const int n = id / DIM;
  const int k = id - n * DIM;
  wT[(size_t)n * DIM + k] = (bf16)vec[idx[(size_t)k * G3 + n]];
}

// ---------------- kernel 3: w_out = x_norm(25600x512) @ W_ih(512x1536) ----------------
// WG (8 waves) owns 128 rows x 64 cols. The 64x32 B panel (4KB) is staged in
// LDS (shared by all 8 waves, killing the 8x redundant global B traffic) and
// double-buffered with async global->LDS copies overlapping the WMMAs.
__global__ __launch_bounds__(256)
void gemm_wout_kernel(const bf16* __restrict__ xn,
                      const bf16* __restrict__ wihT,
                      float* __restrict__ wout) {
  __shared__ __align__(64) bf16 bsm[2][64 * 32];   // 2 x 4KB B panels

  const int tid  = threadIdx.x;
  const int wave = tid >> 5;
  const int lane = tid & 31;
  const int mb   = blockIdx.x % (NROWS / 128);     // 0..199 (128-row block)
  const int nc   = blockIdx.x / (NROWS / 128);     // 0..23  (64-col chunk)

  const int m = lane & 15;
  const int h = lane >> 4;
  const int mt = mb * 8 + wave;                    // this wave's 16-row tile

  const bf16* arow = xn + (size_t)(mt * 16 + m) * DIM;

  // per-thread copy slice: col = tid/4 (0..63), 16B segment = tid%4
  const int ccol = tid >> 2;
  const int cseg = tid & 3;
  const bf16* gsrc = wihT + (size_t)(nc * 64 + ccol) * DIM + cseg * 8;
  bf16* ldst0 = &bsm[0][ccol * 32 + cseg * 8];
  bf16* ldst1 = &bsm[1][ccol * 32 + cseg * 8];

  // prologue: stage K-step 0
  lds_copy16(gsrc, ldst0);
  lds_copy_fence();
  __syncthreads();

  v8f acc[4] = {};
  for (int kk = 0; kk < 16; ++kk) {
    // kick off next panel while computing on the current one
    if (kk < 15) lds_copy16(gsrc + (kk + 1) * 32, (kk & 1) ? ldst0 : ldst1);

    v16bf a = *(const v16bf*)(arow + kk * 32 + 16 * h);

    const bf16* bbase = &bsm[kk & 1][0];
#pragma unroll
    for (int nt = 0; nt < 4; ++nt) {
      v16bf b = *(const v16bf*)(bbase + (nt * 16 + m) * 32 + 16 * h);
      acc[nt] = wmma_bf16(a, b, acc[nt]);
    }

    lds_copy_fence();   // own async copy landed
    __syncthreads();    // everyone's slice of the next panel landed
  }

#pragma unroll
  for (int nt = 0; nt < 4; ++nt) {
    const int col = nc * 64 + nt * 16 + m;
#pragma unroll
    for (int j = 0; j < 8; ++j) {
      const int row = mt * 16 + 8 * h + j;
      wout[(size_t)row * G3 + col] = acc[nt][j];
    }
  }
}

// ---------------- kernel 4: persistent GRU recurrence ----------------
// 48 WGs x 128 thr. WG w owns u columns [32w, 32w+32). W_hh slice lives in LDS
// for the whole kernel. One device barrier per step (u double-buffered), then
// every WG redundantly computes the full gate+LN update into its private LDS h.
__global__ __launch_bounds__(THR_REC)
void gru_rec_kernel(const float* __restrict__ wout,
                    const bf16*  __restrict__ whhT,
                    const float* __restrict__ gamma,
                    const float* __restrict__ beta,
                    float* __restrict__ u_buf,
                    float* __restrict__ out,
                    unsigned* __restrict__ bar) {
  extern __shared__ char smem[];
  bf16*  wlds = (bf16*)(smem);            // [32 cols][512 K]  32KB
  bf16*  hb   = (bf16*)(smem + 32768);    // h as bf16 (GEMM A) 32KB
  float* hf   = (float*)(smem + 65536);   // h as f32           64KB
  float* hp   = (float*)(smem + 131072);  // pre-LN h           64KB
  float* stat = (float*)(smem + 196608);  // mean[32], scale[32]

  const int tid  = threadIdx.x;
  const int lane = tid & 31;
  const int wave = tid >> 5;
  const int wg   = blockIdx.x;
  const int colbase = wg * 32;

  // cache W_hh slice (transposed: wlds[n][k] = W_hh[k][colbase+n]); zero h0
  for (int i = tid; i < 32 * DIM; i += THR_REC) {
    wlds[i] = whhT[(size_t)colbase * DIM + i];
    hf[i] = 0.f;
    hb[i] = (bf16)0.f;
  }
  __syncthreads();

  const int mt = wave & 1;        // M tile: rows 0-15 / 16-31
  const int nt = wave >> 1;       // N tile: local cols 0-15 / 16-31
  const int m  = lane & 15;
  const int h  = lane >> 4;

  for (int t = 0; t < T_STEPS; ++t) {
    float* u = u_buf + (size_t)(t & 1) * BATCH * G3;

    // --- GEMM: u_slice = h(32x512) @ W_hh[:, colbase:colbase+32] ---
    {
      const bf16* arow = hb   + (size_t)(mt * 16 + m) * DIM;
      const bf16* brow = wlds + (size_t)(nt * 16 + m) * DIM;
      v8f acc = {};
      for (int kk = 0; kk < 16; ++kk) {
        v16bf a = *(const v16bf*)(arow + kk * 32 + 16 * h);
        v16bf b = *(const v16bf*)(brow + kk * 32 + 16 * h);
        acc = wmma_bf16(a, b, acc);
      }
      const int col = colbase + nt * 16 + m;
#pragma unroll
      for (int j = 0; j < 8; ++j) {
        const int row = mt * 16 + 8 * h + j;
        u[(size_t)row * G3 + col] = acc[j];
      }
    }

    // --- device-scope barrier: all u slices visible ---
    __threadfence();
    __syncthreads();
    if (tid == 0) {
      __hip_atomic_fetch_add(bar, 1u, __ATOMIC_ACQ_REL, __HIP_MEMORY_SCOPE_AGENT);
      const unsigned target = (unsigned)(t + 1) * NWG_REC;
      while (__hip_atomic_load(bar, __ATOMIC_ACQUIRE, __HIP_MEMORY_SCOPE_AGENT) < target)
        __builtin_amdgcn_s_sleep(8);
    }
    __syncthreads();

    // --- gates (redundant per WG): hp = z*h + (1-z)*tanh(wh + uh*r) ---
    const float* wo = wout + (size_t)t * BATCH * G3;
    for (int i = tid; i < BATCH * DIM; i += THR_REC) {
      const int r = i >> 9;
      const int g = i & (DIM - 1);
      const float* ur = u  + (size_t)r * G3;
      const float* wr = wo + (size_t)r * G3;
      const float uh = ur[g], uz = ur[g + DIM], uur = ur[g + 2 * DIM];
      const float wh = wr[g], wz = wr[g + DIM], wwr = wr[g + 2 * DIM];
      const float zt = 1.f / (1.f + __expf(-(wz + uz)));
      const float rt = 1.f / (1.f + __expf(-(wwr + uur)));
      const float hc = tanhf(wh + uh * rt);
      hp[i] = zt * hf[i] + (1.f - zt) * hc;
    }
    __syncthreads();

    // --- LN stats: wave handles 8 rows, lane-strided reduce over 512 ---
    for (int rr = 0; rr < 8; ++rr) {
      const int r = wave * 8 + rr;
      float s = 0.f, ss = 0.f;
#pragma unroll
      for (int ii = 0; ii < 16; ++ii) {
        const float v = hp[r * DIM + lane + 32 * ii];
        s += v; ss += v * v;
      }
#pragma unroll
      for (int mm = 1; mm < 32; mm <<= 1) {
        s  += __shfl_xor(s,  mm);
        ss += __shfl_xor(ss, mm);
      }
      if (lane == 0) {
        const float mean = s * (1.0f / DIM);
        float var = (ss - (float)DIM * mean * mean) * (1.0f / (DIM - 1));
        var = fmaxf(var, 0.f);
        stat[r]      = mean;
        stat[32 + r] = 1.0f / (sqrtf(var) + EPS_F);
      }
    }
    __syncthreads();

    // --- normalize; refresh private h; WG0 writes output h[t] ---
    for (int i = tid; i < BATCH * DIM; i += THR_REC) {
      const int r = i >> 9;
      const int g = i & (DIM - 1);
      const float v = gamma[g] * (hp[i] - stat[r]) * stat[32 + r] + beta[g];
      hf[i] = v;
      hb[i] = (bf16)v;
      if (wg == 0) out[(size_t)t * BATCH * DIM + i] = v;
    }
    __syncthreads();
  }
}

// ---------------- host ----------------
extern "C" void kernel_launch(void* const* d_in, const int* in_sizes, int n_in,
                              void* d_out, int out_size, void* d_ws, size_t ws_size,
                              hipStream_t stream) {
  const float* x     = (const float*)d_in[0];
  const float* vih   = (const float*)d_in[1];
  const float* vhh   = (const float*)d_in[2];
  const float* g0    = (const float*)d_in[3];
  const float* b0    = (const float*)d_in[4];
  const float* gamma = (const float*)d_in[5];
  const float* beta  = (const float*)d_in[6];
  const int*   iih   = (const int*)d_in[7];
  const int*   ihh   = (const int*)d_in[8];
  float* out = (float*)d_out;

  char* ws = (char*)d_ws;
  float*    wout  = (float*)(ws + OFF_WOUT);
  bf16*     xn    = (bf16*) (ws + OFF_XN);
  bf16*     wihT  = (bf16*) (ws + OFF_WIH);
  bf16*     whhT  = (bf16*) (ws + OFF_WHH);
  float*    u_buf = (float*)(ws + OFF_U);
  unsigned* bar   = (unsigned*)(ws + OFF_BAR);

  bar_init_kernel<<<1, 1, 0, stream>>>(bar);
  ln_x_kernel<<<NROWS / 8, 256, 0, stream>>>(x, g0, b0, xn);

  const int gather_elems = G3 * DIM;
  gather_wT_kernel<<<(gather_elems + 255) / 256, 256, 0, stream>>>(vih, iih, wihT);
  gather_wT_kernel<<<(gather_elems + 255) / 256, 256, 0, stream>>>(vhh, ihh, whhT);

  // 200 M-blocks x 24 64-col chunks = 4800 WGs, 8 waves each
  gemm_wout_kernel<<<4800, 256, 0, stream>>>(xn, wihT, wout);

  const size_t rec_lds = 196608 + 64 * sizeof(float);
  gru_rec_kernel<<<NWG_REC, THR_REC, rec_lds, stream>>>(
      wout, whhT, gamma, beta, u_buf, out, bar);
}
